// SGA_35356170780799
// MI455X (gfx1250) — compile-verified, matrix operand order checked by reading, not run on hardware
//
#include <hip/hip_runtime.h>
#include <hip/hip_bf16.h>

// Problem constants (from reference): B=8, C=256, d=32, H=W=128, N=16384
#define BATCH 8
#define CCH   256
#define DCH   32
#define NTOK  16384
#define NT    128          // tokens per workgroup tile
#define TILES (NTOK / NT)  // 128 tiles per batch

typedef __attribute__((ext_vector_type(16))) __bf16 v16bf;
typedef __attribute__((ext_vector_type(8)))  float  v8f;

// branchless softplus: no EXEC-mask divergence around WMMA regions
__device__ __forceinline__ float softplusf(float x) {
    return fmaxf(x, 0.0f) + log1pf(expf(-fabsf(x)));
}

// Fragment mappings (16-bit WMMA, wave32):
//  A 16x32: lane = h*16 + (m&15); elem e<8 -> K=h*8+e ; e>=8 -> K=16+h*8+(e-8)
//  B 32x16: lane = h*16 + (n&15); same K mapping
//  inverse (h,e) from k_local kl: h=(kl>>3)&1 ; e=(kl&7)+((kl>>4)<<3)
//  C/D 16x16 f32: elem j -> row m = j + 8*(lane>>4), col n = lane&15

__global__ __launch_bounds__(256) void SGA_zero_ws(float* p, int n) {
    int i = blockIdx.x * 256 + threadIdx.x;
    if (i < n) p[i] = 0.0f;
}

// stage a [mtc*16 x 256] f32 row-major weight block as A-fragments (bf16)
__device__ __forceinline__ void stage_weights_A(
    __bf16* dst, const float* __restrict__ W, int row_base, int mtc, int tid)
{
    for (int i = tid; i < mtc * 8 * 32; i += 256) {
        int ln = i & 31, kt = (i >> 5) & 7, mt = i >> 8;
        int h = ln >> 4;
        int m = row_base + mt * 16 + (ln & 15);
        int c0 = kt * 32 + h * 8;
        const float* wr = W + (size_t)m * CCH + c0;
        float4 a0 = *(const float4*)(wr);
        float4 a1 = *(const float4*)(wr + 4);
        float4 a2 = *(const float4*)(wr + 16);
        float4 a3 = *(const float4*)(wr + 20);
        v16bf f;
        f[0]=(__bf16)a0.x; f[1]=(__bf16)a0.y; f[2]=(__bf16)a0.z; f[3]=(__bf16)a0.w;
        f[4]=(__bf16)a1.x; f[5]=(__bf16)a1.y; f[6]=(__bf16)a1.z; f[7]=(__bf16)a1.w;
        f[8]=(__bf16)a2.x; f[9]=(__bf16)a2.y; f[10]=(__bf16)a2.z; f[11]=(__bf16)a2.w;
        f[12]=(__bf16)a3.x; f[13]=(__bf16)a3.y; f[14]=(__bf16)a3.z; f[15]=(__bf16)a3.w;
        *(v16bf*)(dst + (size_t)i * 16) = f;   // [mt][kt][lane][16]
    }
}

// stage x tile [256 x NT] (channel-major, token-contiguous) as B-fragments
__device__ __forceinline__ void stage_x_B(
    __bf16 (*xB)[8][32][16], const float* __restrict__ xb, int n0, int tid)
{
    for (int i = tid; i < CCH * (NT / 4); i += 256) {
        int c  = i >> 5;          // 0..255 (NT/4 == 32)
        int n4 = i & 31;
        int n  = n4 * 4;
        float4 vx = *(const float4*)&xb[(size_t)c * NTOK + n0 + n];
        int kt = c >> 5;
        int cl = c & 31;
        int h  = (cl >> 3) & 1;
        int e  = (cl & 7) + ((cl >> 4) << 3);
        float vv[4] = {vx.x, vx.y, vx.z, vx.w};
#pragma unroll
        for (int t = 0; t < 4; ++t) {
            int nn = n + t;
            xB[kt][nn >> 4][h * 16 + (nn & 15)][e] = (__bf16)vv[t];
        }
    }
}

// ---------------- Kernel A: k = softplus(Wk x + bk); k_sum; attn += k v^T ----
__global__ __launch_bounds__(256) void SGA_kernelA(
    const float* __restrict__ x,  const float* __restrict__ Wk,
    const float* __restrict__ bk, const float* __restrict__ Wv,
    const float* __restrict__ bv, float* __restrict__ attn_ws,
    float* __restrict__ ksum_ws)
{
    __shared__ __bf16 xB [8][8][32][16];  // x tile, B-frag: K=C(256), N=tok(128)  64KB
    __shared__ __bf16 WkA[2][8][32][16];  // Wk, A-frag: M=d(32), K=C(256)         16KB
    __shared__ __bf16 kA [2][4][32][16];  // k,  A-frag: M=d(32), K=tok(128)        8KB
    __shared__ __bf16 WvA[4][8][32][16];  // Wv strip, A-frag: M=64, K=C(256)      32KB
    __shared__ __bf16 vB [4][4][32][16];  // v^T strip, B-frag: K=tok(128), N=64   16KB

    const int tid  = threadIdx.x;
    const int lane = tid & 31;
    const int wave = tid >> 5;
    const int b    = blockIdx.x >> 7;          // TILES == 128
    const int n0   = (blockIdx.x & 127) * NT;
    const float* xb = x + (size_t)b * CCH * NTOK;

    stage_x_B(xB, xb, n0, tid);
    stage_weights_A(&WkA[0][0][0][0], Wk, 0, 2, tid);
    // prefetch first Wv strip while k GEMM runs (lowers to global_prefetch_b8)
    for (int i = tid; i < 512; i += 256)
        __builtin_prefetch((const char*)Wv + (size_t)i * 128, 0, 0);
    __syncthreads();

    // k GEMM: [32 x 128] = Wk[32x256] @ x[256x128]; wave w owns token-tile nt=w
    {
        const int nt = wave;
        const int h = lane >> 4, tok_lo = lane & 15;
        for (int mt = 0; mt < 2; ++mt) {
            v8f acc = {};
            for (int kt = 0; kt < 8; ++kt) {
                v16bf a  = *(const v16bf*)&WkA[mt][kt][lane][0];
                v16bf bb = *(const v16bf*)&xB [kt][nt][lane][0];
                acc = __builtin_amdgcn_wmma_f32_16x16x32_bf16(
                        false, a, false, bb, (short)0, acc, false, false);
            }
            for (int j = 0; j < 8; ++j) {
                int m = mt * 16 + j + 8 * h;                 // d-row
                float kv = softplusf(acc[j] + bk[m]);
                int ktok = nt * 16 + tok_lo;
                int kl   = ktok & 31;
                int ha   = (kl >> 3) & 1;
                int ea   = (kl & 7) + ((kl >> 4) << 3);
                kA[m >> 4][ktok >> 5][ha * 16 + (m & 15)][ea] = (__bf16)kv;
                float s = kv;                 // token reduction in 16-lane half
                s += __shfl_xor(s, 1, 32);
                s += __shfl_xor(s, 2, 32);
                s += __shfl_xor(s, 4, 32);
                s += __shfl_xor(s, 8, 32);
                if (tok_lo == 0) atomicAdd(&ksum_ws[b * DCH + m], s);
            }
        }
    }
    __syncthreads();

    // 4 strips of 64 v-channels: v = Wv x + bv, then attn += k v^T
    for (int s = 0; s < 4; ++s) {
        stage_weights_A(&WvA[0][0][0][0], Wv, s * 64, 4, tid);
        if (s < 3) {   // prefetch next strip (64*256 f32 = 512 cachelines)
            const char* nw = (const char*)(Wv + (size_t)(s + 1) * 64 * CCH);
            for (int i = tid; i < 512; i += 256)
                __builtin_prefetch(nw + (size_t)i * 128, 0, 0);
        }
        __syncthreads();

        // v strip GEMM: [64 x 128]; wave w owns token-tile nt=w, 4 row tiles
        {
            const int nt = wave;
            const int h = lane >> 4, tok_lo = lane & 15;
            for (int mt = 0; mt < 4; ++mt) {
                v8f acc = {};
                for (int kt = 0; kt < 8; ++kt) {
                    v16bf a  = *(const v16bf*)&WvA[mt][kt][lane][0];
                    v16bf bb = *(const v16bf*)&xB [kt][nt][lane][0];
                    acc = __builtin_amdgcn_wmma_f32_16x16x32_bf16(
                            false, a, false, bb, (short)0, acc, false, false);
                }
                for (int j = 0; j < 8; ++j) {
                    int el = mt * 16 + j + 8 * h;            // channel in strip
                    float vv = acc[j] + bv[s * 64 + el];
                    int ktok = nt * 16 + tok_lo;
                    int kl   = ktok & 31;
                    int hb   = (kl >> 3) & 1;
                    int eb   = (kl & 7) + ((kl >> 4) << 3);
                    vB[ktok >> 5][el >> 4][hb * 16 + (el & 15)][eb] = (__bf16)vv;
                }
            }
        }
        __syncthreads();

        // attn strip GEMM: [32 x 64] += kA[32x128] @ vB[128x64]
        {
            const int mt = wave >> 2, nt = wave & 3;
            const int h = lane >> 4, e_lo = lane & 15;
            v8f acc = {};
            for (int kt = 0; kt < 4; ++kt) {
                v16bf a  = *(const v16bf*)&kA[mt][kt][lane][0];
                v16bf bb = *(const v16bf*)&vB[kt][nt][lane][0];
                acc = __builtin_amdgcn_wmma_f32_16x16x32_bf16(
                        false, a, false, bb, (short)0, acc, false, false);
            }
            for (int j = 0; j < 8; ++j) {
                int d   = mt * 16 + j + 8 * h;
                int ech = s * 64 + nt * 16 + e_lo;
                atomicAdd(&attn_ws[((size_t)b * DCH + d) * CCH + ech], acc[j]);
            }
        }
        __syncthreads();
    }
}

// ---------------- Kernel B: q = softplus(Wq x + bq); out = attn^T q^T * norm --
__global__ __launch_bounds__(256) void SGA_kernelB(
    const float* __restrict__ x,  const float* __restrict__ Wq,
    const float* __restrict__ bq, const float* __restrict__ attn_ws,
    const float* __restrict__ ksum_ws, float* __restrict__ out)
{
    __shared__ __bf16 xB  [8][8][32][16]; // x tile B-frag                       64KB
    __shared__ __bf16 WqA [2][8][32][16]; // Wq A-frag                           16KB
    __shared__ __bf16 attnA[16][32][16];  // attn^T A-frag: M=C(256), K=d(32)    16KB
    __shared__ __bf16 qB  [8][32][16];    // q^T  B-frag: K=d(32), N=tok(128)     8KB
    __shared__ float  normL[NT];
    __shared__ float  ksumL[DCH];

    const int tid  = threadIdx.x;
    const int lane = tid & 31;
    const int wave = tid >> 5;
    const int b    = blockIdx.x >> 7;
    const int n0   = (blockIdx.x & 127) * NT;
    const float* xb = x + (size_t)b * CCH * NTOK;

    stage_x_B(xB, xb, n0, tid);
    stage_weights_A(&WqA[0][0][0][0], Wq, 0, 2, tid);
    // stage attn^T as A-fragments (rows of attn[d][:] are contiguous)
    for (int i = tid; i < DCH * (CCH / 4); i += 256) {
        int d   = i >> 6;
        int ech = (i & 63) * 4;
        float4 av = *(const float4*)&attn_ws[((size_t)b * DCH + d) * CCH + ech];
        int h = (d >> 3) & 1;
        int e = (d & 7) + ((d >> 4) << 3);
        float vv[4] = {av.x, av.y, av.z, av.w};
#pragma unroll
        for (int t = 0; t < 4; ++t) {
            int ec = ech + t;
            attnA[ec >> 4][h * 16 + (ec & 15)][e] = (__bf16)vv[t];
        }
    }
    if (tid < DCH) ksumL[tid] = ksum_ws[b * DCH + tid] + 1e-6f;  // EPS
    __syncthreads();

    // q GEMM + norm: wave w owns token-tile nt=w
    {
        const int nt = wave;
        const int h = lane >> 4, tok_lo = lane & 15;
        float denom = 0.0f;
        for (int mt = 0; mt < 2; ++mt) {
            v8f acc = {};
            for (int kt = 0; kt < 8; ++kt) {
                v16bf a  = *(const v16bf*)&WqA[mt][kt][lane][0];
                v16bf bb = *(const v16bf*)&xB [kt][nt][lane][0];
                acc = __builtin_amdgcn_wmma_f32_16x16x32_bf16(
                        false, a, false, bb, (short)0, acc, false, false);
            }
            for (int j = 0; j < 8; ++j) {
                int d = mt * 16 + j + 8 * h;
                float qv = softplusf(acc[j] + bq[d]);
                denom += qv * ksumL[d];
                int hb = (d >> 3) & 1;
                int eb = (d & 7) + ((d >> 4) << 3);
                qB[nt][hb * 16 + tok_lo][eb] = (__bf16)qv;
            }
        }
        denom += __shfl_xor(denom, 16, 32);   // combine both d-halves per token
        if (lane < 16) normL[nt * 16 + lane] = 1.0f / denom;
    }
    __syncthreads();

    // out GEMM: [256 x 128] = attn^T[256x32] @ q^T[32x128]; K=32 -> 1 WMMA/tile
    {
        const int nt = wave;
        const int h = lane >> 4, tok_lo = lane & 15;
        const float nrm = normL[nt * 16 + tok_lo];
        v16bf bb = *(const v16bf*)&qB[nt][lane][0];
        float* ob = out + (size_t)b * CCH * NTOK + n0 + nt * 16 + tok_lo;
        for (int mt = 0; mt < 16; ++mt) {
            v8f acc = {};
            v16bf a = *(const v16bf*)&attnA[mt][lane][0];
            acc = __builtin_amdgcn_wmma_f32_16x16x32_bf16(
                    false, a, false, bb, (short)0, acc, false, false);
#pragma unroll
            for (int j = 0; j < 8; ++j) {
                int ech = mt * 16 + j + 8 * h;
                ob[(size_t)ech * NTOK] = acc[j] * nrm;
            }
        }
    }
}

extern "C" void kernel_launch(void* const* d_in, const int* in_sizes, int n_in,
                              void* d_out, int out_size, void* d_ws, size_t ws_size,
                              hipStream_t stream) {
    (void)in_sizes; (void)n_in; (void)out_size; (void)ws_size;
    const float* x  = (const float*)d_in[0];
    const float* Wq = (const float*)d_in[1];
    const float* bq = (const float*)d_in[2];
    const float* Wk = (const float*)d_in[3];
    const float* bk = (const float*)d_in[4];
    const float* Wv = (const float*)d_in[5];
    const float* bv = (const float*)d_in[6];
    float* out = (float*)d_out;

    float* attn_ws = (float*)d_ws;                      // [B][32][256]
    float* ksum_ws = attn_ws + BATCH * DCH * CCH;       // [B][32]
    const int ws_elems = BATCH * DCH * CCH + BATCH * DCH;

    SGA_zero_ws<<<(ws_elems + 255) / 256, 256, 0, stream>>>(attn_ws, ws_elems);
    SGA_kernelA<<<BATCH * TILES, 256, 0, stream>>>(x, Wk, bk, Wv, bv, attn_ws, ksum_ws);
    SGA_kernelB<<<BATCH * TILES, 256, 0, stream>>>(x, Wq, bq, attn_ws, ksum_ws, out);
}
